// ResidualBlock_85779086836318
// MI455X (gfx1250) — matrix operand order repaired and checked by reading, not compiled, not run on hardware
//
#include <hip/hip_runtime.h>

typedef __attribute__((ext_vector_type(16))) _Float16 v16h;
typedef __attribute__((ext_vector_type(8)))  _Float16 v8h;
typedef __attribute__((ext_vector_type(8)))  float    v8f;
typedef unsigned int u32x4 __attribute__((ext_vector_type(4)));
typedef int          i32x8 __attribute__((ext_vector_type(8)));
typedef int          i32x4 __attribute__((ext_vector_type(4)));

#define BATCH_ 8
#define SEQ_   2048
#define DM_    1024
#define E_     2048
#define NST_   16
#define KC_    4
#define R_     64
#define MR_    (BATCH_*SEQ_)   /* 16384 rows of flattened (b,l) */

#define KSTEP_ 64              /* K elements staged per double-buffer stage     */
#define ROWH_  72              /* padded LDS row pitch in halves (144B)         */
#define TILEB_ (128*ROWH_*2)   /* bytes per staged tile (18432)                 */

#if defined(__has_builtin)
#if __has_builtin(__builtin_amdgcn_tensor_load_to_lds) && __has_builtin(__builtin_amdgcn_s_wait_tensorcnt)
#define HAVE_TDM 1
#endif
#endif

union V16U { v16h v; v8h h[2]; };

__device__ __forceinline__ float sigmoidf_(float x) { return 1.f / (1.f + __expf(-x)); }

#if defined(HAVE_TDM)
// Issue one 2D TDM tile load: KSTEP_ x 128 elements (2B each), row stride
// strideEl elements in memory, into LDS at byte offset ldsOff with 16B of
// padding appended after every 128B row (-> 144B LDS pitch).
// D# packing per CDNA5 ISA (groups 2,3 zero = 2D).
__device__ __forceinline__ void tdm_load_2d(unsigned long long gaddr, unsigned int ldsOff,
                                            unsigned int tenH, unsigned int strideEl) {
  u32x4 g0;
  g0[0] = 1u;                                            // count=1, user descriptor
  g0[1] = ldsOff;                                        // lds_addr (bytes)
  g0[2] = (unsigned int)(gaddr & 0xffffffffu);           // global_addr[31:0]
  g0[3] = (unsigned int)((gaddr >> 32) & 0x1ffffffu)     // global_addr[56:32]
        | (2u << 30);                                    // type=2 ("image")
  i32x8 g1;
  g1[0] = (int)((1u << 16)     /* data_size=1 -> 2 bytes            */
        | (1u << 20)           /* pad_enable                        */
        | (4u << 22)           /* pad_interval: 32 dwords (128B)    */
        | (3u << 25));         /* pad_amount: 4 dwords (16B)        */
  g1[1] = (int)((KSTEP_ & 0xffffu) << 16);               // tensor_dim0 lo16 (=64)
  g1[2] = (int)((tenH & 0xffffu) << 16);                 // dim0 hi16=0 | tensor_dim1 lo16
  g1[3] = (int)(((tenH >> 16) & 0xffffu) | ((unsigned)KSTEP_ << 16)); // dim1 hi | tile_dim0
  g1[4] = (int)(128u);                                   // tile_dim1=128, tile_dim2=0
  g1[5] = (int)strideEl;                                 // tensor_dim0_stride lo32
  g1[6] = 0;                                             // stride0 hi16 | stride1 lo16
  g1[7] = 0;                                             // stride1 hi32
  i32x4 gz4 = {0, 0, 0, 0};
  i32x8 gz8 = {0, 0, 0, 0, 0, 0, 0, 0};
  __builtin_amdgcn_tensor_load_to_lds(g0, g1, gz4, gz4, gz8, 0);
}
#endif

// ---------------------------------------------------------------- f32 -> f16
__global__ __launch_bounds__(256)
void cvt_f16_kernel(const float* __restrict__ s, _Float16* __restrict__ d, int n) {
  int i = blockIdx.x * 256 + threadIdx.x;
  if (i < n) d[i] = (_Float16)s[i];
}

// ---------------------------------------------------------------- RMSNorm -> f16
__global__ __launch_bounds__(256)
void rmsnorm_kernel(const float* __restrict__ x, const float* __restrict__ w,
                    _Float16* __restrict__ h) {
  const int row = blockIdx.x;
  const float* xr = x + (size_t)row * DM_;
  float s = 0.f;
#pragma unroll
  for (int i = 0; i < DM_ / 256; ++i) { float v = xr[threadIdx.x + i * 256]; s += v * v; }
#pragma unroll
  for (int o = 16; o > 0; o >>= 1) s += __shfl_down(s, o, 32);
  __shared__ float red[8];
  if ((threadIdx.x & 31) == 0) red[threadIdx.x >> 5] = s;
  __syncthreads();
  float tot = 0.f;
#pragma unroll
  for (int i = 0; i < 8; ++i) tot += red[i];
  const float inv = rsqrtf(tot * (1.f / DM_) + 1e-5f);
#pragma unroll
  for (int i = 0; i < DM_ / 256; ++i) {
    int c = threadIdx.x + i * 256;
    h[(size_t)row * DM_ + c] = (_Float16)(xr[c] * inv * w[c]);
  }
}

// ---------------------------------------------------------------- WMMA GEMM
// out[m,n] = sum_k A[m,k] * W[n,k]   (A: MxK f16 row-major, W: NxK f16 row-major)
// Block = 8 waves = 128x128 tile; wave = 64x32 (4x2 wmma frags); K staged in
// 64-wide double-buffered LDS stages fed by the Tensor Data Mover (wave 0
// issues 2 descriptors/stage, waits TENSORcnt, workgroup barrier releases).
enum { EPI_F16 = 0, EPI_XPROJ = 1, EPI_SOFTPLUS = 2, EPI_RESID = 3 };

template <int EPI>
__global__ __launch_bounds__(256)
void gemm_wmma_kernel(const _Float16* __restrict__ A, const _Float16* __restrict__ W,
                      int M, int N, int K,
                      float* __restrict__ outF, _Float16* __restrict__ outH,
                      _Float16* __restrict__ outH2,
                      const float* __restrict__ bias, const float* __restrict__ resid) {
  __shared__ _Float16 lds[2][2][128 * ROWH_];   // [buf][A/B][row*ROWH_+col], 72KB

  const int lane = threadIdx.x & 31;
  const int wave = threadIdx.x >> 5;
  const int mBlk = blockIdx.x * 128;
  const int nBlk = blockIdx.y * 128;
  const int m0l = (wave >> 2) * 64;             // wave's M offset inside block tile
  const int n0l = (wave & 3) * 32;              // wave's N offset inside block tile
  const bool active = (nBlk + n0l < N);         // N always multiple of 32
  const int lr  = lane & 15;
  const bool hi = lane >= 16;
  const unsigned int nRows = (unsigned int)(((N - nBlk) < 128) ? (N - nBlk) : 128);

  v8f acc[4][2];
#pragma unroll
  for (int i = 0; i < 4; ++i)
#pragma unroll
    for (int j = 0; j < 2; ++j)
#pragma unroll
      for (int r = 0; r < 8; ++r) acc[i][j][r] = 0.f;

  const int nk = K / KSTEP_;

  // ---- prologue: stage 0 into buffer 0
#if defined(HAVE_TDM)
  if (wave == 0) {
    tdm_load_2d((unsigned long long)(uintptr_t)(A + (size_t)mBlk * K), 0 * TILEB_, 128,   (unsigned)K);
    tdm_load_2d((unsigned long long)(uintptr_t)(W + (size_t)nBlk * K), 1 * TILEB_, nRows, (unsigned)K);
  }
#else
  for (int c = threadIdx.x; c < 1024; c += 256) {   // 128 rows x 8 chunks of 16B
    const int row = c >> 3, seg = c & 7;
    *(v8h*)&lds[0][0][(size_t)row * ROWH_ + seg * 8] =
        *(const v8h*)(A + (size_t)(mBlk + row) * K + seg * 8);
    int wrow = nBlk + row; if (wrow >= N) wrow = N - 1;
    *(v8h*)&lds[0][1][(size_t)row * ROWH_ + seg * 8] =
        *(const v8h*)(W + (size_t)wrow * K + seg * 8);
  }
#endif

  int cur = 0;
  for (int s = 0; s < nk; ++s) {
#if defined(HAVE_TDM)
    if (wave == 0) __builtin_amdgcn_s_wait_tensorcnt(0);
#endif
    __syncthreads();                              // stage `cur` visible to all waves

    // ---- issue next stage into the other buffer
    if (s + 1 < nk) {
      const size_t k = (size_t)(s + 1) * KSTEP_;
      const int nb = cur ^ 1;
#if defined(HAVE_TDM)
      if (wave == 0) {
        tdm_load_2d((unsigned long long)(uintptr_t)(A + (size_t)mBlk * K + k),
                    (unsigned)(nb * 2 + 0) * TILEB_, 128, (unsigned)K);
        tdm_load_2d((unsigned long long)(uintptr_t)(W + (size_t)nBlk * K + k),
                    (unsigned)(nb * 2 + 1) * TILEB_, nRows, (unsigned)K);
      }
#else
      for (int c = threadIdx.x; c < 1024; c += 256) {
        const int row = c >> 3, seg = c & 7;
        *(v8h*)&lds[nb][0][(size_t)row * ROWH_ + seg * 8] =
            *(const v8h*)(A + (size_t)(mBlk + row) * K + k + seg * 8);
        int wrow = nBlk + row; if (wrow >= N) wrow = N - 1;
        *(v8h*)&lds[nb][1][(size_t)row * ROWH_ + seg * 8] =
            *(const v8h*)(W + (size_t)wrow * K + k + seg * 8);
      }
#endif
    }

    // ---- consume stage `cur`
    if (active) {
      const _Float16* As = &lds[cur][0][0];
      const _Float16* Bs = &lds[cur][1][0];
#pragma unroll
      for (int ks = 0; ks < 2; ++ks) {           // two K=32 sub-steps per stage
        V16U a[4], b[2];
#pragma unroll
        for (int i = 0; i < 4; ++i) {            // A frag: lanes0-15 K{0-7,16-23}, lanes16-31 +8
          const _Float16* p = As + (size_t)(m0l + i * 16 + lr) * ROWH_ + ks * 32 + (hi ? 8 : 0);
          a[i].h[0] = *(const v8h*)p;
          a[i].h[1] = *(const v8h*)(p + 16);
        }
#pragma unroll
        for (int j = 0; j < 2; ++j) {            // B frag: lanes0-15 K{0-7,16-23}, lanes16-31 +8
          const _Float16* p = Bs + (size_t)(n0l + j * 16 + lr) * ROWH_ + ks * 32 + (hi ? 8 : 0);
          b[j].h[0] = *(const v8h*)p;
          b[j].h[1] = *(const v8h*)(p + 16);
        }
#pragma unroll
        for (int i = 0; i < 4; ++i)
#pragma unroll
          for (int j = 0; j < 2; ++j)
            acc[i][j] = __builtin_amdgcn_wmma_f32_16x16x32_f16(
                false, a[i].v, false, b[j].v, (short)0, acc[i][j], false, false);
      }
    }
    cur ^= 1;
  }

  if (!active) return;
  // C/D layout: VGPR r -> lanes 0-15: (M=r, N=lane); lanes 16-31: (M=r+8, N=lane-16)
#pragma unroll
  for (int i = 0; i < 4; ++i) {
#pragma unroll
    for (int j = 0; j < 2; ++j) {
#pragma unroll
      for (int r = 0; r < 8; ++r) {
        const int m = mBlk + m0l + i * 16 + (hi ? r + 8 : r);
        const int n = nBlk + n0l + j * 16 + lr;
        const float v = acc[i][j][r];
        if (EPI == EPI_F16) {
          outH[(size_t)m * N + n] = (_Float16)v;
        } else if (EPI == EPI_XPROJ) {
          outF[(size_t)m * N + n] = v;
          if (n < R_) outH2[(size_t)m * R_ + n] = (_Float16)v;
        } else if (EPI == EPI_SOFTPLUS) {
          const float t = v + bias[n];
          outF[(size_t)m * N + n] = (t > 20.f) ? t : __logf(1.f + __expf(t));
        } else {  // EPI_RESID
          outF[(size_t)m * N + n] = v + resid[(size_t)m * N + n];
        }
      }
    }
  }
}

// -------------------------------------------- causal depthwise conv (K=4) + SiLU
__global__ __launch_bounds__(256)
void conv_silu_kernel(const _Float16* __restrict__ xz, const float* __restrict__ cw,
                      const float* __restrict__ cb, _Float16* __restrict__ u) {
  const size_t gid = (size_t)blockIdx.x * 256 + threadIdx.x;
  const int e = (int)(gid & (E_ - 1));
  const size_t m = gid >> 11;                 // b*SEQ + t
  const int t = (int)(m & (SEQ_ - 1));
  const size_t brow = m - t;                  // b*SEQ
  float acc = cb[e];
#pragma unroll
  for (int j = 0; j < KC_; ++j) {
    const int tt = t - (KC_ - 1) + j;
    if (tt >= 0) acc += (float)xz[(brow + tt) * (size_t)(2 * E_) + e] * cw[e * KC_ + j];
  }
  u[gid] = (_Float16)(acc * sigmoidf_(acc));
}

// -------------------------------------------- selective scan (sequential in t)
__global__ __launch_bounds__(256)
void scan_kernel(float* __restrict__ deltaYs, const _Float16* __restrict__ u,
                 const float* __restrict__ dbc, const float* __restrict__ A_log) {
  const int gid  = blockIdx.x * 256 + threadIdx.x;
  const int lane = threadIdx.x & 31;
  const int e = gid & (E_ - 1);
  const int b = gid >> 11;
  float A[NST_], st[NST_];
#pragma unroll
  for (int n = 0; n < NST_; ++n) { A[n] = -__expf(A_log[e * NST_ + n]); st[n] = 0.f; }
  const size_t rowBase = (size_t)b * SEQ_;
  for (int t = 0; t < SEQ_; ++t) {
    const size_t m = rowBase + t;
    const float bc = dbc[m * 96 + 64 + lane];   // one coalesced B||C load per wave
    const size_t ue = m * (size_t)E_ + e;
    const float d  = deltaYs[ue];
    const float du = d * (float)u[ue];
    float y = 0.f;
#pragma unroll
    for (int n = 0; n < NST_; ++n) {
      const float Bn = __shfl(bc, n, 32);
      const float Cn = __shfl(bc, n + 16, 32);
      st[n] = st[n] * __expf(d * A[n]) + du * Bn;
      y += st[n] * Cn;
    }
    deltaYs[ue] = y;
  }
}

// -------------------------------------------- y = (ys + u*D) * silu(z) -> f16
__global__ __launch_bounds__(256)
void gate_kernel(const float* __restrict__ ys, const _Float16* __restrict__ u,
                 const _Float16* __restrict__ xz, const float* __restrict__ Dskip,
                 _Float16* __restrict__ yg) {
  const size_t gid = (size_t)blockIdx.x * 256 + threadIdx.x;
  const int e = (int)(gid & (E_ - 1));
  const size_t m = gid >> 11;
  const float z = (float)xz[m * (size_t)(2 * E_) + E_ + e];
  const float y = (ys[gid] + (float)u[gid] * Dskip[e]) * (z * sigmoidf_(z));
  yg[gid] = (_Float16)y;
}

// ----------------------------------------------------------------- host side
static inline size_t align256(size_t x) { return (x + 255) & ~(size_t)255; }

extern "C" void kernel_launch(void* const* d_in, const int* in_sizes, int n_in,
                              void* d_out, int out_size, void* d_ws, size_t ws_size,
                              hipStream_t stream) {
  (void)in_sizes; (void)n_in; (void)out_size; (void)ws_size;
  const float* x         = (const float*)d_in[0];
  const float* norm_w    = (const float*)d_in[1];
  const float* in_proj_w = (const float*)d_in[2];
  const float* conv_w    = (const float*)d_in[3];
  const float* conv_b    = (const float*)d_in[4];
  const float* x_proj_w  = (const float*)d_in[5];
  const float* dt_proj_w = (const float*)d_in[6];
  const float* dt_proj_b = (const float*)d_in[7];
  const float* A_log     = (const float*)d_in[8];
  const float* D_skip    = (const float*)d_in[9];
  const float* out_proj_w= (const float*)d_in[10];
  float* out = (float*)d_out;

  char* ws = (char*)d_ws;
  size_t off = 0;
  auto alloc = [&](size_t bytes) { char* p = ws + off; off = align256(off + bytes); return p; };

  _Float16* h16    = (_Float16*)alloc((size_t)MR_ * DM_ * 2);       //  32 MB
  _Float16* Win16  = (_Float16*)alloc((size_t)2 * E_ * DM_ * 2);    //   8 MB
  _Float16* Wx16   = (_Float16*)alloc((size_t)96 * E_ * 2);         // 384 KB
  _Float16* Wdt16  = (_Float16*)alloc((size_t)E_ * R_ * 2);         // 256 KB
  _Float16* Wout16 = (_Float16*)alloc((size_t)DM_ * E_ * 2);        //   4 MB
  _Float16* xz16   = (_Float16*)alloc((size_t)MR_ * 2 * E_ * 2);    // 128 MB
  _Float16* u16    = (_Float16*)alloc((size_t)MR_ * E_ * 2);        //  64 MB
  float*    dbc32  = (float*)   alloc((size_t)MR_ * 96 * 4);        //   6 MB
  _Float16* dt16   = (_Float16*)alloc((size_t)MR_ * R_ * 2);        //   2 MB
  float*    dy32   = (float*)   alloc((size_t)MR_ * E_ * 4);        // 128 MB (delta, then y)
  _Float16* yg16   = (_Float16*)alloc((size_t)MR_ * E_ * 2);        //  64 MB

  // 1. weight conversions to f16
  {
    int n;
    n = 2 * E_ * DM_; cvt_f16_kernel<<<(n + 255) / 256, 256, 0, stream>>>(in_proj_w,  Win16,  n);
    n = 96 * E_;      cvt_f16_kernel<<<(n + 255) / 256, 256, 0, stream>>>(x_proj_w,   Wx16,   n);
    n = E_ * R_;      cvt_f16_kernel<<<(n + 255) / 256, 256, 0, stream>>>(dt_proj_w,  Wdt16,  n);
    n = DM_ * E_;     cvt_f16_kernel<<<(n + 255) / 256, 256, 0, stream>>>(out_proj_w, Wout16, n);
  }

  // 2. RMSNorm -> h16
  rmsnorm_kernel<<<MR_, 256, 0, stream>>>(x, norm_w, h16);

  // 3. in_proj: xz16 = h16 @ Win16^T   (16384 x 4096 x 1024)
  gemm_wmma_kernel<EPI_F16><<<dim3(MR_ / 128, (2 * E_) / 128), 256, 0, stream>>>(
      h16, Win16, MR_, 2 * E_, DM_, nullptr, xz16, nullptr, nullptr, nullptr);

  // 4. depthwise causal conv + SiLU -> u16
  conv_silu_kernel<<<(MR_ * E_) / 256, 256, 0, stream>>>(xz16, conv_w, conv_b, u16);

  // 5. x_proj: dbc32 = u16 @ Wx16^T   (16384 x 96 x 2048); dt part also -> dt16
  gemm_wmma_kernel<EPI_XPROJ><<<dim3(MR_ / 128, 1), 256, 0, stream>>>(
      u16, Wx16, MR_, 96, E_, dbc32, nullptr, dt16, nullptr, nullptr);

  // 6. dt_proj + softplus: dy32 = softplus(dt16 @ Wdt16^T + b)  (16384 x 2048 x 64)
  gemm_wmma_kernel<EPI_SOFTPLUS><<<dim3(MR_ / 128, E_ / 128), 256, 0, stream>>>(
      dt16, Wdt16, MR_, E_, R_, dy32, nullptr, nullptr, dt_proj_b, nullptr);

  // 7. selective scan: rewrites dy32 in place with y
  scan_kernel<<<(BATCH_ * E_) / 256, 256, 0, stream>>>(dy32, u16, dbc32, A_log);

  // 8. gating -> yg16
  gate_kernel<<<(MR_ * E_) / 256, 256, 0, stream>>>(dy32, u16, xz16, D_skip, yg16);

  // 9. out_proj + residual: out = yg16 @ Wout16^T + x  (16384 x 1024 x 2048)
  gemm_wmma_kernel<EPI_RESID><<<dim3(MR_ / 128, DM_ / 128), 256, 0, stream>>>(
      yg16, Wout16, MR_, DM_, E_, out, nullptr, nullptr, nullptr, x);
}
